// GPT_88673894793833
// MI455X (gfx1250) — compile-verified, matrix operand order checked by reading, not run on hardware
//
#include <hip/hip_runtime.h>
#include <math.h>

// ---------------------------------------------------------------------------
// Problem constants (match reference)
// ---------------------------------------------------------------------------
#define NB    4          // batch
#define NT    827        // sequence length
#define NC    1024       // channels
#define NH    16         // heads
#define HD    64         // head dim
#define HW    256        // region size
#define TPAD  832        // NT padded to multiple of 32 (26*32)

typedef __bf16 bf16_t;
typedef __attribute__((ext_vector_type(8)))  bf16_t bhalf8;
typedef __attribute__((ext_vector_type(16))) bf16_t bhalf16;
typedef __attribute__((ext_vector_type(8)))  float  vf8;

__device__ __forceinline__ int imin(int a, int b) { return a < b ? a : b; }

__device__ __forceinline__ bhalf16 combine8(bhalf8 lo, bhalf8 hi) {
    bhalf16 r;
#pragma unroll
    for (int i = 0; i < 8; ++i) { r[i] = lo[i]; r[i + 8] = hi[i]; }
    return r;
}

// Load one lane's 16 bf16 elements of an A operand tile for
// v_wmma_f32_16x16x32_bf16 straight from global memory.  Lane groups 0-15 /
// 16-31 hold the two K-halves: each lane reads two contiguous 16-byte chunks
// (K = khalf*8 + [0..7] and K = 16 + khalf*8 + [0..7]).
__device__ __forceinline__ bhalf16 load_ab(const bf16_t* __restrict__ rowk, int khalf) {
    bhalf8 lo = *(const bhalf8*)(rowk + khalf * 8);
    bhalf8 hi = *(const bhalf8*)(rowk + 16 + khalf * 8);
    return combine8(lo, hi);
}

// Async global->LDS stage of one 16-byte chunk (CDNA5 GLOBAL_LOAD_ASYNC_TO_LDS,
// GVS addressing: SGPR base + 32-bit VGPR byte offset).  Tracked by ASYNCcnt.
__device__ __forceinline__ void async_copy16(unsigned lds_addr, unsigned gbyte_off,
                                             const void* base) {
    asm volatile("global_load_async_to_lds_b128 %0, %1, %2"
                 :: "v"(lds_addr), "v"(gbyte_off),
                    "s"((unsigned long long)(uintptr_t)base)
                 : "memory");
}
__device__ __forceinline__ void async_wait_all() {
    asm volatile("s_wait_asynccnt 0x0" ::: "memory");
}

// ---------------------------------------------------------------------------
// fp32 -> bf16 convert
// ---------------------------------------------------------------------------
__global__ void cvt_bf16_kernel(const float* __restrict__ src, bf16_t* __restrict__ dst, int n) {
    int i = blockIdx.x * blockDim.x + threadIdx.x;
    if (i < n) dst[i] = (bf16_t)src[i];
}

// ---------------------------------------------------------------------------
// Generic WMMA GEMM:  out[M,N] = A[M,K] @ W[N,K]^T + bias
// block = 128 threads (4 waves); block computes 64x64.  The shared W tile
// (64 cols x 32 K, 4KB bf16) is staged in LDS via async DMA, double-buffered,
// and consumed by all 4 waves (4x less vector-memory traffic on W).
// grid = (ceil(M/64), N/64)
// ---------------------------------------------------------------------------
template <bool OUTF32>
__global__ void gemm_wmma_kernel(const bf16_t* __restrict__ A, const bf16_t* __restrict__ W,
                                 const float* __restrict__ bias, void* __restrict__ outp,
                                 int M, int N, int K) {
    __shared__ bf16_t wt[2][64][32];
    const int lane  = threadIdx.x & 31;
    const int wave  = threadIdx.x >> 5;
    const int mtile = blockIdx.x * 4 + wave;
    const int ncol0 = blockIdx.y * 64;
    const int m15   = lane & 15;
    const int khalf = lane >> 4;
    const int arow  = imin(mtile * 16 + m15, M - 1);
    const bf16_t* Ar = A + (size_t)arow * K;

    auto stage = [&](int buf, int k0) {
#pragma unroll
        for (int i = 0; i < 2; ++i) {
            int c  = (int)threadIdx.x + 128 * i;   // 256 chunks of 16B = 4KB
            int n  = c >> 2;
            int ko = (c & 3) * 8;
            unsigned goff  = (unsigned)(((size_t)imin(ncol0 + n, N - 1) * K + k0 + ko) * 2);
            unsigned laddr = (unsigned)(uintptr_t)&wt[buf][n][ko];
            async_copy16(laddr, goff, W);
        }
    };

    stage(0, 0);
    vf8 acc[4] = {};
    for (int k0 = 0; k0 < K; k0 += 32) {
        int buf = (k0 >> 5) & 1;
        async_wait_all();
        __syncthreads();
        if (k0 + 32 < K) stage(buf ^ 1, k0 + 32);
        bhalf16 a = load_ab(Ar + k0, khalf);
#pragma unroll
        for (int j = 0; j < 4; ++j) {
            int nl = j * 16 + m15;
            bhalf8 lo = *(const bhalf8*)&wt[buf][nl][khalf * 8];
            bhalf8 hi = *(const bhalf8*)&wt[buf][nl][16 + khalf * 8];
            acc[j] = __builtin_amdgcn_wmma_f32_16x16x32_bf16(
                false, a, false, combine8(lo, hi), (short)0, acc[j], false, false);
        }
        __syncthreads();
    }
#pragma unroll
    for (int j = 0; j < 4; ++j) {
        int col = ncol0 + j * 16 + m15;
        if (col >= N) continue;
        float bv = bias ? bias[col] : 0.f;
#pragma unroll
        for (int r = 0; r < 8; ++r) {
            int row = mtile * 16 + khalf * 8 + r;   // C/D layout: M = r + 8*(lane>>4)
            if (row < M) {
                float v = acc[j][r] + bv;
                if (OUTF32) ((float*)outp)[(size_t)row * N + col] = v;
                else        ((bf16_t*)outp)[(size_t)row * N + col] = (bf16_t)v;
            }
        }
    }
}

// ---------------------------------------------------------------------------
// att[b,h,qi,kj] = scale * sum_d q*k + h[b,0,qi,kj]   (stored bf16, TPAD stride)
// K-token tile (64 tokens x 32 dims) staged in LDS, shared by the 4 M-waves.
// grid = (ceil(NT/64)=13, 13, B*NH), block 128
// ---------------------------------------------------------------------------
__global__ void qk_kernel(const bf16_t* __restrict__ q, const bf16_t* __restrict__ k,
                          const float* __restrict__ h, bf16_t* __restrict__ att, float scale) {
    __shared__ bf16_t wt[2][64][32];
    const int bh = blockIdx.z, b = bh >> 4, hh = bh & 15;
    const int lane = threadIdx.x & 31, wave = threadIdx.x >> 5;
    const int mtile = blockIdx.x * 4 + wave;
    const int ncol0 = blockIdx.y * 64;
    const int m15   = lane & 15;
    const int khalf = lane >> 4;
    const int arow  = imin(mtile * 16 + m15, NT - 1);
    const bf16_t* Aq = q + ((size_t)(b * NT + arow)) * NC + hh * HD;

    auto stage = [&](int buf, int k0) {
#pragma unroll
        for (int i = 0; i < 2; ++i) {
            int c  = (int)threadIdx.x + 128 * i;
            int n  = c >> 2;
            int ko = (c & 3) * 8;
            int tok = imin(ncol0 + n, NT - 1);
            unsigned goff  = (unsigned)((((size_t)(b * NT + tok)) * NC + hh * HD + k0 + ko) * 2);
            unsigned laddr = (unsigned)(uintptr_t)&wt[buf][n][ko];
            async_copy16(laddr, goff, k);
        }
    };

    stage(0, 0);
    vf8 acc[4] = {};
    for (int k0 = 0; k0 < HD; k0 += 32) {
        int buf = (k0 >> 5) & 1;
        async_wait_all();
        __syncthreads();
        if (k0 + 32 < HD) stage(buf ^ 1, k0 + 32);
        bhalf16 a = load_ab(Aq + k0, khalf);
#pragma unroll
        for (int j = 0; j < 4; ++j) {
            int nl = j * 16 + m15;
            bhalf8 lo = *(const bhalf8*)&wt[buf][nl][khalf * 8];
            bhalf8 hi = *(const bhalf8*)&wt[buf][nl][16 + khalf * 8];
            acc[j] = __builtin_amdgcn_wmma_f32_16x16x32_bf16(
                false, a, false, combine8(lo, hi), (short)0, acc[j], false, false);
        }
        __syncthreads();
    }
#pragma unroll
    for (int j = 0; j < 4; ++j) {
        int col = ncol0 + j * 16 + m15;
        if (col >= NT) continue;
#pragma unroll
        for (int r = 0; r < 8; ++r) {
            int row = mtile * 16 + khalf * 8 + r;
            if (row < NT) {
                float v = acc[j][r] * scale + h[((size_t)b * NT + row) * NT + col];
                att[((size_t)bh * NT + row) * TPAD + col] = (bf16_t)v;
            }
        }
    }
}

// ---------------------------------------------------------------------------
// Block reductions (blockDim.x == 256)
// ---------------------------------------------------------------------------
__device__ __forceinline__ float block_reduce(float v, float* red, bool is_max) {
    int tid = threadIdx.x;
    red[tid] = v;
    __syncthreads();
    for (int s = 128; s > 0; s >>= 1) {
        if (tid < s) red[tid] = is_max ? fmaxf(red[tid], red[tid + s]) : (red[tid] + red[tid + s]);
        __syncthreads();
    }
    float r = red[0];
    __syncthreads();
    return r;
}

// ---------------------------------------------------------------------------
// Region softmax over 256 cols for R1/R2/R3.  grid=(256 rows, 64 bh, 3 regions)
// ---------------------------------------------------------------------------
__global__ void region_softmax_kernel(bf16_t* __restrict__ att) {
    __shared__ float red[256];
    const int row0[3] = {285, 571, 571};
    const int col0[3] = {0, 0, 286};
    int r = blockIdx.x, bh = blockIdx.y, reg = blockIdx.z;
    int row = row0[reg] + r, col = col0[reg] + threadIdx.x;
    bf16_t* p = att + ((size_t)bh * NT + row) * TPAD + col;
    float x = (float)*p;
    float M = block_reduce(x, red, true);
    float e = __expf(x - M);
    float S = block_reduce(e, red, false);
    *p = (bf16_t)(e / S);
}

// ---------------------------------------------------------------------------
// bi_epi_ratio[b,r] = 256 * sum_{h,c}(reg3*mask) / (maskcnt * sum_{h,c} reg3)
// Must run after region softmax, before region multiplies.
// grid=(256 rows, B), block 256
// ---------------------------------------------------------------------------
__global__ void stats_kernel(const bf16_t* __restrict__ att, const float* __restrict__ f12,
                             const float* __restrict__ b12, float* __restrict__ ratio) {
    __shared__ float red[256];
    int r = blockIdx.x, b = blockIdx.y, c = threadIdx.x;
    float m = (f12[((size_t)b * HW + r) * HW + c] * b12[((size_t)b * HW + c) * HW + r] >= 0.1f)
                  ? 1.f : 0.f;
    float ts = 0.f, tm = 0.f;
#pragma unroll 4
    for (int hh = 0; hh < NH; ++hh) {
        float v = (float)att[(((size_t)(b * NH + hh)) * NT + 571 + r) * TPAD + 286 + c];
        ts += v;
        tm += v * m;
    }
    float MC = block_reduce(m, red, false);
    float TS = block_reduce(ts, red, false);
    float TM = block_reduce(tm, red, false);
    if (threadIdx.x == 0) ratio[b * HW + r] = (256.f * TM) / (MC * TS);
}

// ---------------------------------------------------------------------------
// att[:,:,rs,cs] *= f[b,i,j] * bmat[b,j,i]  (per region)
// grid=(256 rows, 64 bh, 3 regions), block 256
// ---------------------------------------------------------------------------
__global__ void region_mul_kernel(bf16_t* __restrict__ att,
                                  const float* __restrict__ f01, const float* __restrict__ b01,
                                  const float* __restrict__ f02, const float* __restrict__ b02,
                                  const float* __restrict__ f12, const float* __restrict__ b12) {
    const int row0[3] = {285, 571, 571};
    const int col0[3] = {0, 0, 286};
    int i = blockIdx.x, bh = blockIdx.y, reg = blockIdx.z, j = threadIdx.x;
    int b = bh >> 4;
    const float* F = (reg == 0) ? f01 : (reg == 1) ? f02 : f12;
    const float* Bm = (reg == 0) ? b01 : (reg == 1) ? b02 : b12;
    float m = F[((size_t)b * HW + i) * HW + j] * Bm[((size_t)b * HW + j) * HW + i];
    size_t idx = ((size_t)bh * NT + row0[reg] + i) * TPAD + col0[reg] + j;
    att[idx] = (bf16_t)((float)att[idx] * m);
}

// ---------------------------------------------------------------------------
// Per-row: stage in LDS; for rows >=571 redo R2/R3 256-wide softmaxes on the
// post-multiply values; causal mask; full softmax; write bf16 with zero pad.
// grid=(NT, 64), block 256
// ---------------------------------------------------------------------------
__global__ void row_softmax_kernel(bf16_t* __restrict__ att) {
    __shared__ float srow[TPAD];
    __shared__ float red[256];
    int qi = blockIdx.x, bh = blockIdx.y, tid = threadIdx.x;
    bf16_t* rowp = att + ((size_t)bh * NT + qi) * TPAD;

    for (int c = tid; c < TPAD; c += 256) srow[c] = (c < NT) ? (float)rowp[c] : 0.f;
    __syncthreads();

    if (qi >= 571) {
        const int bases[2] = {0, 286};
#pragma unroll
        for (int ri = 0; ri < 2; ++ri) {
            int idx = bases[ri] + tid;
            float x = srow[idx];
            float M = block_reduce(x, red, true);
            float e = __expf(x - M);
            float S = block_reduce(e, red, false);
            srow[idx] = e / S;
            __syncthreads();
        }
    }

    // causal full softmax over 827 cols
    float mx = -INFINITY;
    for (int c = tid; c < NT; c += 256)
        if (c <= qi) mx = fmaxf(mx, srow[c]);
    float M = block_reduce(mx, red, true);
    float s = 0.f;
    for (int c = tid; c < NT; c += 256)
        if (c <= qi) s += __expf(srow[c] - M);
    float S = block_reduce(s, red, false);
    for (int c = tid; c < TPAD; c += 256) {
        float o = (c < NT && c <= qi) ? (__expf(srow[c] - M) / S) : 0.f;
        rowp[c] = (bf16_t)o;
    }
}

// ---------------------------------------------------------------------------
// vT[b,h,d,k] = v[b*NT+k, h*64+d]  (zero-padded k >= NT)
// ---------------------------------------------------------------------------
__global__ void vtrans_kernel(const bf16_t* __restrict__ v, bf16_t* __restrict__ vt) {
    size_t i = (size_t)blockIdx.x * blockDim.x + threadIdx.x;
    size_t total = (size_t)NB * NH * HD * TPAD;
    if (i >= total) return;
    int kk = i % TPAD;
    size_t t = i / TPAD;
    int d = t % HD;
    size_t bh = t / HD;
    int b = (int)(bh >> 4), hh = (int)(bh & 15);
    vt[i] = (kk < NT) ? v[((size_t)b * NT + kk) * NC + hh * HD + d] : (bf16_t)0.f;
}

// ---------------------------------------------------------------------------
// y2[b*NT+row, h*64+d] = sum_k aw[b,h,row,k] * vT[b,h,d,k]
// vT tile (64 d x 32 k) staged in LDS, shared by the 4 M-waves.
// grid=(13, 64 bh), block 128; wave does 16 rows x 64 cols (all of HD)
// ---------------------------------------------------------------------------
__global__ void av_kernel(const bf16_t* __restrict__ aw, const bf16_t* __restrict__ vt,
                          bf16_t* __restrict__ y2) {
    __shared__ bf16_t wt[2][64][32];
    const int bh = blockIdx.y, b = bh >> 4, hh = bh & 15;
    const int lane = threadIdx.x & 31, wave = threadIdx.x >> 5;
    const int mtile = blockIdx.x * 4 + wave;
    const int m15 = lane & 15;
    const int khalf = lane >> 4;
    const int arow = imin(mtile * 16 + m15, NT - 1);
    const bf16_t* Ar = aw + ((size_t)bh * NT + arow) * TPAD;

    auto stage = [&](int buf, int k0) {
#pragma unroll
        for (int i = 0; i < 2; ++i) {
            int c  = (int)threadIdx.x + 128 * i;
            int n  = c >> 2;              // d index 0..63
            int ko = (c & 3) * 8;
            unsigned goff  = (unsigned)((((size_t)bh * HD + n) * TPAD + k0 + ko) * 2);
            unsigned laddr = (unsigned)(uintptr_t)&wt[buf][n][ko];
            async_copy16(laddr, goff, vt);
        }
    };

    stage(0, 0);
    vf8 acc[4] = {};
    for (int k0 = 0; k0 < TPAD; k0 += 32) {
        int buf = (k0 >> 5) & 1;
        async_wait_all();
        __syncthreads();
        if (k0 + 32 < TPAD) stage(buf ^ 1, k0 + 32);
        bhalf16 a = load_ab(Ar + k0, khalf);
#pragma unroll
        for (int j = 0; j < 4; ++j) {
            int nl = j * 16 + m15;
            bhalf8 lo = *(const bhalf8*)&wt[buf][nl][khalf * 8];
            bhalf8 hi = *(const bhalf8*)&wt[buf][nl][16 + khalf * 8];
            acc[j] = __builtin_amdgcn_wmma_f32_16x16x32_bf16(
                false, a, false, combine8(lo, hi), (short)0, acc[j], false, false);
        }
        __syncthreads();
    }
#pragma unroll
    for (int j = 0; j < 4; ++j) {
        int d = j * 16 + m15;
#pragma unroll
        for (int r = 0; r < 8; ++r) {
            int row = mtile * 16 + khalf * 8 + r;
            if (row < NT)
                y2[((size_t)b * NT + row) * NC + hh * HD + d] = (bf16_t)acc[j][r];
        }
    }
}

// ---------------------------------------------------------------------------
// Launcher
// ---------------------------------------------------------------------------
extern "C" void kernel_launch(void* const* d_in, const int* in_sizes, int n_in,
                              void* d_out, int out_size, void* d_ws, size_t ws_size,
                              hipStream_t stream) {
    const float* x   = (const float*)d_in[0];
    const float* xkv = (const float*)d_in[1];
    const float* h   = (const float*)d_in[2];
    const float* f01 = (const float*)d_in[3];
    const float* f02 = (const float*)d_in[4];
    const float* f12 = (const float*)d_in[5];
    const float* b01 = (const float*)d_in[6];
    const float* b02 = (const float*)d_in[7];
    const float* b12 = (const float*)d_in[8];
    const float* Wq  = (const float*)d_in[9];
    const float* bq  = (const float*)d_in[10];
    const float* Wk  = (const float*)d_in[11];
    const float* bk  = (const float*)d_in[12];
    const float* Wv  = (const float*)d_in[13];
    const float* bv  = (const float*)d_in[14];
    const float* Wp  = (const float*)d_in[15];
    const float* bp  = (const float*)d_in[16];
    float* out = (float*)d_out;

    const size_t XB  = (size_t)NB * NT * NC;   // 3,387,392
    const size_t CC  = (size_t)NC * NC;        // 1,048,576
    const size_t ATT = (size_t)NB * NH * NT * TPAD;
    const size_t VTN = (size_t)NB * NH * HD * TPAD;

    char* ws = (char*)d_ws;
    size_t off = 0;
    auto alloc = [&](size_t bytes) -> void* {
        void* p = ws + off;
        off += (bytes + 255) & ~(size_t)255;
        return p;
    };
    bf16_t* xb   = (bf16_t*)alloc(XB * 2);
    bf16_t* xkvb = (bf16_t*)alloc(XB * 2);
    bf16_t* Wqb  = (bf16_t*)alloc(CC * 2);
    bf16_t* Wkb  = (bf16_t*)alloc(CC * 2);
    bf16_t* Wvb  = (bf16_t*)alloc(CC * 2);
    bf16_t* Wpb  = (bf16_t*)alloc(CC * 2);
    bf16_t* qb   = (bf16_t*)alloc(XB * 2);
    bf16_t* kb   = (bf16_t*)alloc(XB * 2);
    bf16_t* vb   = (bf16_t*)alloc(XB * 2);
    bf16_t* attb = (bf16_t*)alloc(ATT * 2);
    bf16_t* vtb  = (bf16_t*)alloc(VTN * 2);
    bf16_t* y2b  = (bf16_t*)alloc(XB * 2);
    (void)ws_size;

    const int M = NB * NT;   // 3308
    const float scale = 1.0f / sqrtf((float)HD);

    // 1. converts
    cvt_bf16_kernel<<<(int)((XB + 255) / 256), 256, 0, stream>>>(x, xb, (int)XB);
    cvt_bf16_kernel<<<(int)((XB + 255) / 256), 256, 0, stream>>>(xkv, xkvb, (int)XB);
    cvt_bf16_kernel<<<(int)((CC + 255) / 256), 256, 0, stream>>>(Wq, Wqb, (int)CC);
    cvt_bf16_kernel<<<(int)((CC + 255) / 256), 256, 0, stream>>>(Wk, Wkb, (int)CC);
    cvt_bf16_kernel<<<(int)((CC + 255) / 256), 256, 0, stream>>>(Wv, Wvb, (int)CC);
    cvt_bf16_kernel<<<(int)((CC + 255) / 256), 256, 0, stream>>>(Wp, Wpb, (int)CC);

    // 2. Q/K/V projections (WMMA, async-LDS staged weights)
    dim3 gg((M + 63) / 64, NC / 64);
    gemm_wmma_kernel<false><<<gg, 128, 0, stream>>>(xb,   Wqb, bq, qb, M, NC, NC);
    gemm_wmma_kernel<false><<<gg, 128, 0, stream>>>(xkvb, Wkb, bk, kb, M, NC, NC);
    gemm_wmma_kernel<false><<<gg, 128, 0, stream>>>(xkvb, Wvb, bv, vb, M, NC, NC);

    // 3. att = q k^T * scale + h
    dim3 gqk((NT + 63) / 64, (TPAD + 63) / 64, NB * NH);
    qk_kernel<<<gqk, 128, 0, stream>>>(qb, kb, h, attb, scale);

    // 4. region softmaxes, stats (reads post-softmax, pre-multiply), multiplies
    region_softmax_kernel<<<dim3(HW, NB * NH, 3), 256, 0, stream>>>(attb);
    stats_kernel<<<dim3(HW, NB), 256, 0, stream>>>(attb, f12, b12, out + XB);
    region_mul_kernel<<<dim3(HW, NB * NH, 3), 256, 0, stream>>>(attb, f01, b01, f02, b02, f12, b12);

    // 5. per-row region-override + causal + full softmax (in place, bf16)
    row_softmax_kernel<<<dim3(NT, NB * NH), 256, 0, stream>>>(attb);

    // 6. y = aw @ v  (via transposed V for K-contiguous WMMA loads)
    vtrans_kernel<<<(int)((VTN + 255) / 256), 256, 0, stream>>>(vb, vtb);
    av_kernel<<<dim3((NT + 63) / 64, NB * NH), 128, 0, stream>>>(attb, vtb, y2b);

    // 7. output projection -> fp32 d_out
    gemm_wmma_kernel<true><<<gg, 128, 0, stream>>>(y2b, Wpb, bp, out, M, NC, NC);
}
